// AdaFNN_89103391522988
// MI455X (gfx1250) — matrix operand order repaired and checked by reading, not compiled, not run on hardware
//
#include <hip/hip_runtime.h>
#include <hip/hip_bf16.h>
#include <stdint.h>

#define MGRID 128
#define JFEAT 80
#define NBASE 2
#define NNET  160
#define HDIM  256
#define BATCH 2048
#define LN_EPS 1e-6f

typedef __attribute__((ext_vector_type(16))) __bf16        v16bf;
typedef __attribute__((ext_vector_type(8)))  float         v8f;
typedef __attribute__((ext_vector_type(4)))  unsigned int  v4u;
typedef __attribute__((ext_vector_type(8)))  int           v8i;
typedef __attribute__((ext_vector_type(4)))  int           v4i;

union BFrag { v16bf v; v4u q[2]; };

__device__ __forceinline__ unsigned short f2bf(float f) {
  __bf16 b = (__bf16)f;                       // native v_cvt_*bf16 (RNE)
  return __builtin_bit_cast(unsigned short, b);
}
__device__ __forceinline__ float bf2f(unsigned short h) {
  return (float)__builtin_bit_cast(__bf16, h);
}
__device__ __forceinline__ float selu_f(float x) {
  const float kScale = 1.0507009873554805f;
  const float kAlpha = 1.6732632423543772f;
  return x > 0.f ? kScale * x : kScale * kAlpha * (expf(x) - 1.f);
}

// Tensor Data Mover: contiguous 1-D tile of `nelem` f32 from global -> LDS.
// D# per CDNA5 ISA: group0 = {count=1, lds_addr, global_addr, type=2},
// group1 = {data_size=4B, tensor_dim0=nelem, tensor_dim1=1, tile_dim0=nelem,
//           tensor_dim0_stride=nelem}.
__device__ __forceinline__ void tdm_load_f32(unsigned lds_off,
                                             const float* gptr,
                                             unsigned nelem) {
  uint64_t ga = (uint64_t)(uintptr_t)gptr;
  v4u g0;
  g0[0] = 1u;                                           // count=1, user mode
  g0[1] = lds_off;                                      // lds_addr (bytes)
  g0[2] = (unsigned)(ga & 0xffffffffu);                 // global_addr[31:0]
  g0[3] = (unsigned)((ga >> 32) & 0x01ffffffu) | (2u << 30);  // addr[56:32]|type=2
  v8i g1;
  g1[0] = (int)(2u << 16);                              // data_size = 4 bytes
  g1[1] = (int)((nelem & 0xffffu) << 16);               // tensor_dim0[15:0]
  g1[2] = (int)(((nelem >> 16) & 0xffffu) | (1u << 16));// dim0 hi | tensor_dim1=1
  g1[3] = (int)((nelem & 0xffffu) << 16);               // tile_dim0
  g1[4] = 0;                                            // tile_dim1/2 unused
  g1[5] = (int)nelem;                                   // tensor_dim0_stride lo
  g1[6] = 0;
  g1[7] = 0;
  v4i gz; gz[0]=0; gz[1]=0; gz[2]=0; gz[3]=0;
#if __has_include(<hip/amd_detail/amd_gfx1250_TDM.h>)
  v8i gz8; gz8[0]=0; gz8[1]=0; gz8[2]=0; gz8[3]=0; gz8[4]=0; gz8[5]=0; gz8[6]=0; gz8[7]=0;
  __builtin_amdgcn_tensor_load_to_lds(g0, g1, gz, gz, gz8, 0);
#else
  __builtin_amdgcn_tensor_load_to_lds(g0, g1, gz, gz, 0);
#endif
}

// -------- Kernel 1: evaluate all 160 basis MLPs on the grid ----------------
// One workgroup per network n. 256 threads = 8 waves; wave w owns rows
// [16w, 16w+16) of the 128x256 activation tile (zA, bf16 in LDS).
// Weight K-chunks (32x256 f32 = 32 KB, contiguous) stream in via the TDM,
// double buffered so the DMA for chunk kc+1 overlaps convert+GEMM of kc.
__global__ void __launch_bounds__(256, 1)
adafnn_basis_kernel(const float* __restrict__ t,
                    const float* __restrict__ W_in,
                    const float* __restrict__ b_in,
                    const float* __restrict__ W_h,
                    const float* __restrict__ b_h,
                    const float* __restrict__ W_out,
                    const float* __restrict__ b_out,
                    const float* __restrict__ ln_alpha,
                    const float* __restrict__ ln_beta,
                    float* __restrict__ bases) {
  extern __shared__ __align__(16) unsigned char smem_raw[];
  unsigned short* zA  = (unsigned short*)smem_raw;                    // 64 KB
  float* Wraw0        = (float*)(smem_raw + 65536);                   // 32 KB
  float* Wraw1        = (float*)(smem_raw + 65536 + 32768);           // 32 KB
  unsigned short* Wb  = (unsigned short*)(smem_raw + 65536 + 65536);  // 16 KB
  float* bias_s       = (float*)(smem_raw + 65536 + 65536 + 16384);   // 256 f32
  float* alpha_s      = bias_s + HDIM;
  float* beta_s       = alpha_s + HDIM;

  const int n    = blockIdx.x;
  const int tid  = threadIdx.x;
  const int lane = tid & 31;
  const int wv   = tid >> 5;
  const int lrow = lane & 15;
  const int half = lane >> 4;

  // ---- input layer: z = t*W_in + b_in; z += LN(z); selu ----
  {
    const int r = tid >> 1;      // grid row, 2 threads per row
    const int p = tid & 1;
    const float tv = t[r];
    const float* wi = W_in + (size_t)n * HDIM;
    const float* bi = b_in + (size_t)n * HDIM;
    float sum = 0.f, sq = 0.f;
    for (int c = p * 128; c < p * 128 + 128; ++c) {
      float z = tv * wi[c] + bi[c];
      sum += z; sq += z * z;
    }
    sum += __shfl_xor(sum, 1, 32);
    sq  += __shfl_xor(sq, 1, 32);
    float mean = sum * (1.f / HDIM);
    float var  = (sq - sum * mean) * (1.f / (HDIM - 1));   // unbiased
    float istd = 1.f / (sqrtf(var) + LN_EPS);
    const float* al = ln_alpha + (size_t)(0 * NNET + n) * HDIM;
    const float* be = ln_beta  + (size_t)(0 * NNET + n) * HDIM;
    for (int c = p * 128; c < p * 128 + 128; ++c) {
      float z = tv * wi[c] + bi[c];
      float y = z + (z - mean) * istd * al[c] + be[c];
      zA[r * HDIM + c] = f2bf(selu_f(y));
    }
  }

  // ---- 3 hidden layers: z' = z @ W + b; z' += LN(z'); selu ----
  for (int layer = 0; layer < 3; ++layer) {
    __syncthreads();   // previous layer done reading params / Wb / raw bufs
    bias_s[tid]  = b_h[(size_t)(layer * NNET + n) * HDIM + tid];
    alpha_s[tid] = ln_alpha[(size_t)((layer + 1) * NNET + n) * HDIM + tid];
    beta_s[tid]  = ln_beta [(size_t)((layer + 1) * NNET + n) * HDIM + tid];

    v8f vzero;
#pragma unroll
    for (int j = 0; j < 8; ++j) vzero[j] = 0.f;
    v8f acc[16];
#pragma unroll
    for (int nt = 0; nt < 16; ++nt) acc[nt] = vzero;

    const float* Wl = W_h + (size_t)(layer * NNET + n) * HDIM * HDIM;

    // prologue: DMA chunk 0 into raw buffer 0
    if (tid < 32)
      tdm_load_f32((unsigned)(uintptr_t)Wraw0, Wl, 32 * HDIM);

    for (int kc = 0; kc < 8; ++kc) {         // K chunks of 32
      float* cur = (kc & 1) ? Wraw1 : Wraw0;
      float* nxt = (kc & 1) ? Wraw0 : Wraw1;
      if (tid < 32)
        __builtin_amdgcn_s_wait_tensorcnt(0);  // chunk kc resident
      __syncthreads();                          // + prev GEMM done with Wb
      if (kc < 7 && tid < 32)                   // overlap next DMA with GEMM
        tdm_load_f32((unsigned)(uintptr_t)nxt,
                     Wl + (size_t)(kc + 1) * 32 * HDIM, 32 * HDIM);
      // convert+transpose cur[32][256] f32 -> Wb[256][32] bf16 (B layout).
      // Thread tid owns column c = tid; its 32 dest halfwords are contiguous,
      // so pack pairs -> v_cvt_pk_bf16_f32 + ds_store_b32.
      {
        unsigned* wrow = (unsigned*)(Wb + tid * 32);
#pragma unroll
        for (int k2 = 0; k2 < 16; ++k2) {
          float a = cur[(2 * k2 + 0) * 256 + tid];
          float b = cur[(2 * k2 + 1) * 256 + tid];
          unsigned pa = (unsigned)f2bf(a);
          unsigned pb = (unsigned)f2bf(b);
          wrow[k2] = pa | (pb << 16);
        }
      }
      __syncthreads();

      // A fragment: row = 16*wv + lrow; lanes<16: K 0-15; lanes>=16: K 16-31
      BFrag af;
      {
        const v4u* zp = (const v4u*)(zA + (size_t)(wv * 16 + lrow) * HDIM + kc * 32);
        af.q[0] = zp[2 * half];
        af.q[1] = zp[2 * half + 1];
      }
      // B fragments software-pipelined one tile ahead of the WMMA
      const unsigned short* wbase = Wb + lrow * 32 + half * 16;
      BFrag bcur, bnext;
      bcur.q[0] = ((const v4u*)wbase)[0];
      bcur.q[1] = ((const v4u*)wbase)[1];
#pragma unroll
      for (int nt = 0; nt < 16; ++nt) {
        if (nt < 15) {
          const v4u* wp = (const v4u*)(wbase + (nt + 1) * 16 * 32);
          bnext.q[0] = wp[0];
          bnext.q[1] = wp[1];
        }
        acc[nt] = __builtin_amdgcn_wmma_f32_16x16x32_bf16(
            false, af.v, false, bcur.v, (short)0, acc[nt], false, false);
        bcur = bnext;
      }
    }

    // bias + in-register LayerNorm (row stats across 16 lanes x 16 tiles)
    float sumj[8], sqj[8];
#pragma unroll
    for (int j = 0; j < 8; ++j) { sumj[j] = 0.f; sqj[j] = 0.f; }
#pragma unroll
    for (int nt = 0; nt < 16; ++nt) {
      float bv = bias_s[nt * 16 + lrow];
#pragma unroll
      for (int j = 0; j < 8; ++j) {
        float v = acc[nt][j] + bv;
        acc[nt][j] = v;
        sumj[j] += v; sqj[j] += v * v;
      }
    }
    float meanj[8], istdj[8];
#pragma unroll
    for (int j = 0; j < 8; ++j) {
#pragma unroll
      for (int m = 1; m <= 8; m <<= 1) {     // stays inside each 16-lane half
        sumj[j] += __shfl_xor(sumj[j], m, 32);
        sqj[j]  += __shfl_xor(sqj[j], m, 32);
      }
      meanj[j] = sumj[j] * (1.f / HDIM);
      float var = (sqj[j] - sumj[j] * meanj[j]) * (1.f / (HDIM - 1));
      istdj[j] = 1.f / (sqrtf(var) + LN_EPS);
    }
#pragma unroll
    for (int nt = 0; nt < 16; ++nt) {
      int col = nt * 16 + lrow;
      float al = alpha_s[col], be = beta_s[col];
#pragma unroll
      for (int j = 0; j < 8; ++j) {
        float z = acc[nt][j];
        float y = z + (z - meanj[j]) * istdj[j] * al + be;
        int row = wv * 16 + half * 8 + j;    // C layout: lanes>=16 hold M=j+8
        zA[row * HDIM + col] = f2bf(selu_f(y));
      }
    }
    // zA rows are wave-private (read & written only by owner wave): no barrier
  }

  __syncthreads();
  // ---- output layer: bases[n][m] = z . W_out + b_out ----
  {
    const int r = tid >> 1;
    const int p = tid & 1;
    const float* wo = W_out + (size_t)n * HDIM;
    float s = 0.f;
    for (int c = p * 128; c < p * 128 + 128; ++c)
      s += bf2f(zA[r * HDIM + c]) * wo[c];
    s += __shfl_xor(s, 1, 32);
    if (p == 0) bases[(size_t)n * MGRID + r] = s + b_out[n];
  }
}

// -------- Kernel 2: trapezoid inner-product score (bandwidth bound) --------
// thread q -> (b = q/J, j = q%J); consecutive threads hit consecutive j so
// x loads coalesce; bases (80 KB) stays hot in L2.
__global__ void __launch_bounds__(256)
adafnn_score_kernel(const float* __restrict__ x,
                    const float* __restrict__ t,
                    const float* __restrict__ bases,
                    float* __restrict__ out) {
  __shared__ float wq[MGRID];
  int tid = threadIdx.x;
  if (tid < MGRID) {
    float tm = t[tid];
    float wm = 0.f;
    if (tid > 0)         wm += (tm - t[tid - 1]) * 0.5f;
    if (tid < MGRID - 1) wm += (t[tid + 1] - tm) * 0.5f;
    wq[tid] = wm;
  }
  __syncthreads();
  int q = blockIdx.x * blockDim.x + tid;
  if (q >= BATCH * JFEAT) return;
  int b = q / JFEAT, j = q % JFEAT;
  const float* xp = x + (size_t)b * MGRID * JFEAT + j;
  const float* g0 = bases + (size_t)(2 * j + 0) * MGRID;
  const float* g1 = bases + (size_t)(2 * j + 1) * MGRID;
  float a0 = 0.f, a1 = 0.f;
#pragma unroll 4
  for (int m = 0; m < MGRID; ++m) {
    float xv = xp[(size_t)m * JFEAT] * wq[m];
    a0 += xv * g0[m];
    a1 += xv * g1[m];
  }
  out[(size_t)b * (JFEAT * NBASE) + 2 * j + 0] = a0;
  out[(size_t)b * (JFEAT * NBASE) + 2 * j + 1] = a1;
}

extern "C" void kernel_launch(void* const* d_in, const int* in_sizes, int n_in,
                              void* d_out, int out_size, void* d_ws, size_t ws_size,
                              hipStream_t stream) {
  const float* x     = (const float*)d_in[0];
  const float* t     = (const float*)d_in[1];
  const float* W_in  = (const float*)d_in[2];
  const float* b_in  = (const float*)d_in[3];
  const float* W_h   = (const float*)d_in[4];
  const float* b_h   = (const float*)d_in[5];
  const float* W_out = (const float*)d_in[6];
  const float* b_out = (const float*)d_in[7];
  const float* ln_a  = (const float*)d_in[8];
  const float* ln_b  = (const float*)d_in[9];
  float* bases = (float*)d_ws;                 // 160*128 f32 = 80 KB scratch

  size_t shmem = 65536 + 65536 + 16384 + 3 * HDIM * sizeof(float);
  adafnn_basis_kernel<<<NNET, 256, shmem, stream>>>(
      t, W_in, b_in, W_h, b_h, W_out, b_out, ln_a, ln_b, bases);
  adafnn_score_kernel<<<(BATCH * JFEAT + 255) / 256, 256, 0, stream>>>(
      x, t, bases, (float*)d_out);
}
